// Multihead_Self_Attention_45810121179711
// MI455X (gfx1250) — matrix operand order repaired and checked by reading, not compiled, not run on hardware
//
#include <hip/hip_runtime.h>

// ---------------------------------------------------------------------------
// MHA for MI455X (gfx1250): fp32 -> bf16, all GEMMs via v_wmma_f32_16x16x32_bf16
// ---------------------------------------------------------------------------

typedef __bf16 bf16_t;
typedef __attribute__((ext_vector_type(16))) __bf16 v16bf;
typedef __attribute__((ext_vector_type(8)))  __bf16 v8bf;
typedef __attribute__((ext_vector_type(8)))  float  v8f;

#define D_MODEL 1024
#define NUM_HEADS 16
#define D_HEAD 64
#define BATCH 4
#define SEQ 2048
#define M_ROWS (BATCH * SEQ)   // 8192

__device__ __forceinline__ v8f wmma_bf16(v16bf a, v16bf b, v8f c) {
    // D = A(16x32 bf16) x B(32x16 bf16) + C(16x16 f32)
    return __builtin_amdgcn_wmma_f32_16x16x32_bf16(
        /*neg_a=*/false, a, /*neg_b=*/false, b,
        /*c_mod=*/(short)0, c, /*reuse_a=*/false, /*reuse_b=*/false);
}

__device__ __forceinline__ bf16_t f2bf(float f) { return (bf16_t)f; }

// A-fragment builder: two 16B chunks -> 16x32 A layout (lane row = lane%16)
__device__ __forceinline__ v16bf load_afrag(const bf16_t* row, int hh) {
    v8bf lo = *(const v8bf*)(row + hh * 16);
    v8bf hi = *(const v8bf*)(row + hh * 16 + 8);
    v16bf a;
#pragma unroll
    for (int i = 0; i < 8; ++i) { a[i] = lo[i]; a[i + 8] = hi[i]; }
    return a;
}

// ---------------------------------------------------------------------------
// fp32 -> bf16 conversion (n is always a multiple of 4*blockDim here)
// ---------------------------------------------------------------------------
__global__ void cvt_f32_to_bf16(const float* __restrict__ in,
                                bf16_t* __restrict__ out, int n) {
    int i = (blockIdx.x * blockDim.x + threadIdx.x) * 4;
    if (i + 3 < n) {
        float4 f = *(const float4*)(in + i);
        out[i + 0] = f2bf(f.x);
        out[i + 1] = f2bf(f.y);
        out[i + 2] = f2bf(f.z);
        out[i + 3] = f2bf(f.w);
    }
}

// ---------------------------------------------------------------------------
// WMMA GEMM: C[8192,1024] = A[8192,1024] * B^T, B stored [N=1024, K=1024].
// 2x4 register-blocked: one 32x64 output tile per wave. 8 WMMAs reuse
// 2 A-frags x 4 B-frags -> 12 b128 loads per 8 WMMAs (1.5 loads/WMMA).
// mode 0: Q  -> bf16 [b,h,s,64], scaled by 1/sqrt(64)
// mode 1: K  -> bf16 [b,h,s,64]
// mode 2: V  -> bf16 [b,h,64,s]  (transposed for PV B-fragments)
// mode 3: O  -> f32  [m,1024]    (final output projection)
// ---------------------------------------------------------------------------
__global__ __launch_bounds__(256) void gemm16_bf16(
    const bf16_t* __restrict__ A, const bf16_t* __restrict__ B,
    void* __restrict__ Cout, int mode, float scale) {

    const int lane = threadIdx.x & 31;
    const int wid  = threadIdx.x >> 5;
    const int l16  = lane & 15;
    const int hh   = lane >> 4;

    const int tile = blockIdx.x * 8 + wid;   // 4096 wave-tiles total
    const int mT = tile >> 4;                // 256 row tiles of 32
    const int nB = tile & 15;                // 16 col blocks of 64

    const bf16_t* arow0 = A + (size_t)(mT * 32 + l16) * D_MODEL;
    const bf16_t* arow1 = arow0 + (size_t)16 * D_MODEL;
    const bf16_t* brow  = B + (size_t)(nB * 64 + l16) * D_MODEL;

    v8f acc[2][4];
#pragma unroll
    for (int mi = 0; mi < 2; ++mi)
#pragma unroll
        for (int ni = 0; ni < 4; ++ni) acc[mi][ni] = (v8f){};

    for (int k0 = 0; k0 < D_MODEL; k0 += 32) {
        v16bf a0 = load_afrag(arow0 + k0, hh);
        v16bf a1 = load_afrag(arow1 + k0, hh);
#pragma unroll
        for (int ni = 0; ni < 4; ++ni) {
            // B fragment (32x16): lane col = l16; K = hh*16 .. hh*16+15 contiguous
            v16bf b = *(const v16bf*)(brow + (size_t)ni * 16 * D_MODEL + k0 + hh * 16);
            acc[0][ni] = wmma_bf16(a0, b, acc[0][ni]);
            acc[1][ni] = wmma_bf16(a1, b, acc[1][ni]);
        }
    }

#pragma unroll
    for (int mi = 0; mi < 2; ++mi) {
#pragma unroll
        for (int ni = 0; ni < 4; ++ni) {
            const int n = nB * 64 + ni * 16 + l16;
            if (mode == 3) {
                float* O = (float*)Cout;
#pragma unroll
                for (int v = 0; v < 8; ++v) {
                    int m = mT * 32 + mi * 16 + v + 8 * hh;  // C row = v+8*(lane/16)
                    O[(size_t)m * D_MODEL + n] = acc[mi][ni][v];
                }
            } else {
                bf16_t* O = (bf16_t*)Cout;
                const int head = n >> 6, d = n & 63;
#pragma unroll
                for (int v = 0; v < 8; ++v) {
                    int m = mT * 32 + mi * 16 + v + 8 * hh;
                    int b = m >> 11, s = m & (SEQ - 1);
                    size_t idx;
                    if (mode == 2)       // V transposed: [b,h,64,s]
                        idx = ((size_t)(b * NUM_HEADS + head) * D_HEAD + d) * SEQ + s;
                    else                 // Q/K: [b,h,s,64]
                        idx = ((size_t)(b * NUM_HEADS + head) * SEQ + s) * D_HEAD + d;
                    O[idx] = f2bf(acc[mi][ni][v] * scale);
                }
            }
        }
    }
}

// ---------------------------------------------------------------------------
// Flash attention: one wave = 16 query rows of one (b,h); causal key chunks
// of 32 (4 score WMMAs + online softmax + LDS transpose + 4 PV WMMAs).
// Q pre-scaled by 1/sqrt(d_head). Output bf16 [b,s,h,64] == [b,s,1024].
// ---------------------------------------------------------------------------
__global__ __launch_bounds__(256) void flash_attn16(
    const bf16_t* __restrict__ Q, const bf16_t* __restrict__ K,
    const bf16_t* __restrict__ Vt, bf16_t* __restrict__ O) {

    __shared__ bf16_t pLds[8][16 * 32];      // 1 KB per wave (P tile staging)

    const int lane = threadIdx.x & 31;
    const int wid  = threadIdx.x >> 5;
    const int l16  = lane & 15;
    const int hh   = lane >> 4;

    const int tile = blockIdx.x * 8 + wid;   // 8192 tiles = 64 (b,h) * 128 qtiles
    const int qt = tile & 127;
    const int bh = tile >> 7;                // b*16 + head
    const int q0 = qt * 16;

    const bf16_t* Qb = Q  + (size_t)bh * SEQ * D_HEAD;
    const bf16_t* Kb = K  + (size_t)bh * SEQ * D_HEAD;
    const bf16_t* Vb = Vt + (size_t)bh * D_HEAD * SEQ;
    bf16_t* pw = &pLds[wid][0];

    // Q A-fragments (reused for every key chunk): rows q0+l16, d split 0..31/32..63
    const bf16_t* qrow = Qb + (size_t)(q0 + l16) * D_HEAD;
    v16bf qa0 = load_afrag(qrow, hh);
    v16bf qa1 = load_afrag(qrow + 32, hh);

    float m_r[8], l_r[8];
    v8f oacc[4] = {};
#pragma unroll
    for (int v = 0; v < 8; ++v) { m_r[v] = -INFINITY; l_r[v] = 0.0f; }

    const int jEnd = q0 + 16;                // causal bound (exclusive)
    for (int j0 = 0; j0 < jEnd; j0 += 32) {
        // ---- scores: two 16-key subtiles, K-dim = 64 over d (2 WMMAs each)
        v8f s0 = {}, s1 = {};
        {
            const bf16_t* krow = Kb + (size_t)(j0 + l16) * D_HEAD;
            v16bf kb0 = *(const v16bf*)(krow + hh * 16);
            v16bf kb1 = *(const v16bf*)(krow + 32 + hh * 16);
            s0 = wmma_bf16(qa0, kb0, s0);
            s0 = wmma_bf16(qa1, kb1, s0);
        }
        {
            const bf16_t* krow = Kb + (size_t)(j0 + 16 + l16) * D_HEAD;
            v16bf kb0 = *(const v16bf*)(krow + hh * 16);
            v16bf kb1 = *(const v16bf*)(krow + 32 + hh * 16);
            s1 = wmma_bf16(qa0, kb0, s1);
            s1 = wmma_bf16(qa1, kb1, s1);
        }

        // ---- causal mask + online softmax (row = q0 + v + 8*hh, col = j0(+16)+l16)
#pragma unroll
        for (int v = 0; v < 8; ++v) {
            const int qr = q0 + v + 8 * hh;
            if (j0 + l16 > qr)      s0[v] = -INFINITY;
            if (j0 + 16 + l16 > qr) s1[v] = -INFINITY;

            float mx = fmaxf(s0[v], s1[v]);
            mx = fmaxf(mx, __shfl_xor(mx, 1, 32));
            mx = fmaxf(mx, __shfl_xor(mx, 2, 32));
            mx = fmaxf(mx, __shfl_xor(mx, 4, 32));
            mx = fmaxf(mx, __shfl_xor(mx, 8, 32));
            const float mnew  = fmaxf(m_r[v], mx);
            const float alpha = __expf(m_r[v] - mnew);   // exp(-inf)=0 on 1st chunk
            m_r[v] = mnew;

            const float p0 = __expf(s0[v] - mnew);
            const float p1 = __expf(s1[v] - mnew);
            s0[v] = p0; s1[v] = p1;

            float rs = p0 + p1;
            rs += __shfl_xor(rs, 1, 32);
            rs += __shfl_xor(rs, 2, 32);
            rs += __shfl_xor(rs, 4, 32);
            rs += __shfl_xor(rs, 8, 32);
            l_r[v] = l_r[v] * alpha + rs;
#pragma unroll
            for (int nt = 0; nt < 4; ++nt) oacc[nt][v] *= alpha;
        }

        // ---- P: C-layout -> A-layout via wave-private LDS tile (16 rows x 32 keys)
#pragma unroll
        for (int v = 0; v < 8; ++v) {
            const int r = v + 8 * hh;
            pw[r * 32 + l16]      = f2bf(s0[v]);
            pw[r * 32 + 16 + l16] = f2bf(s1[v]);
        }
        asm volatile("s_wait_dscnt 0" ::: "memory");
        v8bf plo = *(const v8bf*)(pw + l16 * 32 + hh * 16);
        v8bf phi = *(const v8bf*)(pw + l16 * 32 + hh * 16 + 8);
        v16bf pa;
#pragma unroll
        for (int i = 0; i < 8; ++i) { pa[i] = plo[i]; pa[i + 8] = phi[i]; }

        // ---- O += P(16x32) x V(32x16) for 4 d-chunks; Vt row d contiguous in j
#pragma unroll
        for (int nt = 0; nt < 4; ++nt) {
            const bf16_t* vrow = Vb + (size_t)(nt * 16 + l16) * SEQ + j0 + hh * 16;
            v16bf vb = *(const v16bf*)vrow;
            oacc[nt] = wmma_bf16(pa, vb, oacc[nt]);
        }
    }

    // ---- normalize and store as [b, s, h*64] (== model layout for O-projection)
    const int b = bh >> 4, head = bh & 15;
#pragma unroll
    for (int v = 0; v < 8; ++v) {
        const float inv = 1.0f / l_r[v];
        const int q = q0 + v + 8 * hh;
        bf16_t* orow = O + ((size_t)(b * SEQ + q) * NUM_HEADS + head) * D_HEAD;
#pragma unroll
        for (int nt = 0; nt < 4; ++nt)
            orow[nt * 16 + l16] = f2bf(oacc[nt][v] * inv);
    }
}

// ---------------------------------------------------------------------------
// Host launch
// ---------------------------------------------------------------------------
extern "C" void kernel_launch(void* const* d_in, const int* in_sizes, int n_in,
                              void* d_out, int out_size, void* d_ws, size_t ws_size,
                              hipStream_t stream) {
    (void)in_sizes; (void)n_in; (void)out_size; (void)ws_size;

    const float* x  = (const float*)d_in[0];
    const float* Wq = (const float*)d_in[1];
    const float* Wk = (const float*)d_in[2];
    const float* Wv = (const float*)d_in[3];
    const float* Wo = (const float*)d_in[4];
    float* out = (float*)d_out;

    char* ws = (char*)d_ws;
    const size_t SZ_X = (size_t)M_ROWS * D_MODEL * sizeof(bf16_t);   // 16 MB
    const size_t SZ_W = (size_t)D_MODEL * D_MODEL * sizeof(bf16_t);  //  2 MB
    bf16_t* xb  = (bf16_t*)(ws);
    bf16_t* wqb = (bf16_t*)(ws + SZ_X);
    bf16_t* wkb = (bf16_t*)(ws + SZ_X + SZ_W);
    bf16_t* wvb = (bf16_t*)(ws + SZ_X + 2 * SZ_W);
    bf16_t* wob = (bf16_t*)(ws + SZ_X + 3 * SZ_W);
    bf16_t* qb  = (bf16_t*)(ws + SZ_X + 4 * SZ_W);
    bf16_t* kb  = (bf16_t*)(ws + 2 * SZ_X + 4 * SZ_W);
    bf16_t* vtb = (bf16_t*)(ws + 3 * SZ_X + 4 * SZ_W);
    bf16_t* ab  = (bf16_t*)(ws + 4 * SZ_X + 4 * SZ_W);               // 88 MB total

    // fp32 -> bf16
    const int nx = M_ROWS * D_MODEL;            // 8388608
    const int nw = D_MODEL * D_MODEL;           // 1048576
    cvt_f32_to_bf16<<<nx / (4 * 256), 256, 0, stream>>>(x,  xb,  nx);
    cvt_f32_to_bf16<<<nw / (4 * 256), 256, 0, stream>>>(Wq, wqb, nw);
    cvt_f32_to_bf16<<<nw / (4 * 256), 256, 0, stream>>>(Wk, wkb, nw);
    cvt_f32_to_bf16<<<nw / (4 * 256), 256, 0, stream>>>(Wv, wvb, nw);
    cvt_f32_to_bf16<<<nw / (4 * 256), 256, 0, stream>>>(Wo, wob, nw);

    // projections: 4096 wave-tiles (32x64 each) / 8 waves per block
    gemm16_bf16<<<512, 256, 0, stream>>>(xb, wqb, qb,  0, 0.125f); // Q * 1/sqrt(64)
    gemm16_bf16<<<512, 256, 0, stream>>>(xb, wkb, kb,  1, 1.0f);
    gemm16_bf16<<<512, 256, 0, stream>>>(xb, wvb, vtb, 2, 1.0f);

    // flash attention: 8192 query tiles / 8 waves per block
    flash_attn16<<<1024, 256, 0, stream>>>(qb, kb, vtb, ab);

    // output projection -> fp32 d_out
    gemm16_bf16<<<512, 256, 0, stream>>>(ab, wob, out, 3, 1.0f);
}
